// Encoder_20298015441662
// MI455X (gfx1250) — matrix operand order repaired and checked by reading, not compiled, not run on hardware
//
#include <hip/hip_runtime.h>
#include <cstdint>
#include <cstddef>

// ---------------------------------------------------------------------------
// MI455X (gfx1250) implementation of the GCN->GRU encoder.
// wave32, WMMA f32_16x16x32_f16 for all dense GEMMs, TDM (tensor_load_to_lds)
// double-buffered B-tile staging, f32 atomics for the edge scatter (the
// memory-bound part at 23.3 TB/s).
// ---------------------------------------------------------------------------

typedef _Float16 v16h __attribute__((ext_vector_type(16)));
typedef _Float16 v8h  __attribute__((ext_vector_type(8)));
typedef float    v8f  __attribute__((ext_vector_type(8)));
typedef unsigned int v4u __attribute__((ext_vector_type(4)));
typedef int          v8i __attribute__((ext_vector_type(8)));
typedef int          v4i __attribute__((ext_vector_type(4)));

#if defined(__HIP_DEVICE_COMPILE__) && __has_builtin(__builtin_amdgcn_tensor_load_to_lds)
#define USE_TDM 1
#else
#define USE_TDM 0
#endif

// ---------------------------- elementwise helpers ---------------------------

__global__ __launch_bounds__(256)
void zero_f32_kernel(float* __restrict__ p, size_t n) {
  size_t i = (size_t)blockIdx.x * blockDim.x + threadIdx.x;
  size_t stride = (size_t)gridDim.x * blockDim.x;
  for (; i < n; i += stride) p[i] = 0.0f;
}

__global__ __launch_bounds__(256)
void cast_f32_f16_kernel(const float* __restrict__ in, _Float16* __restrict__ out, size_t n) {
  size_t i = (size_t)blockIdx.x * blockDim.x + threadIdx.x;
  if (i < n) out[i] = (_Float16)in[i];
}

// W is [K][Nout] row-major (x @ W convention); WT becomes [Nout][K] f16 so the
// WMMA B-fragment loads are contiguous in K.
__global__ __launch_bounds__(256)
void transpose_cast_kernel(const float* __restrict__ W, _Float16* __restrict__ WT,
                           int K, int Nout) {
  int idx = blockIdx.x * blockDim.x + threadIdx.x;
  if (idx < K * Nout) {
    int k = idx / Nout;
    int n = idx - k * Nout;
    WT[n * K + k] = (_Float16)W[idx];
  }
}

// ------------------------------- GCN degree ---------------------------------

__global__ __launch_bounds__(256)
void degree_kernel(const int* __restrict__ dst, float* __restrict__ deg, int E) {
  int i = blockIdx.x * blockDim.x + threadIdx.x;
  if (i < E) atomicAdd(&deg[dst[i]], 1.0f);
}

__global__ __launch_bounds__(256)
void dinv_kernel(const float* __restrict__ deg, float* __restrict__ dinv, int n) {
  int i = blockIdx.x * blockDim.x + threadIdx.x;
  if (i < n) dinv[i] = rsqrtf(deg[i] + 1.0f);  // +1 = self loop; always > 0
}

// ------------------------------ edge scatter --------------------------------
// 64 threads per edge, 4 channels each (float4 gather + 4 f32 atomics).
// agg[dst] += h[src] * dinv[src] * dinv[dst]

__global__ __launch_bounds__(256)
void scatter_kernel(const int* __restrict__ src, const int* __restrict__ dst,
                    const float* __restrict__ h, const float* __restrict__ dinv,
                    float* __restrict__ agg, int E) {
  size_t tid = (size_t)blockIdx.x * blockDim.x + threadIdx.x;
  int e = (int)(tid >> 6);
  if (e >= E) return;
  int c = ((int)tid & 63) << 2;
  int s = src[e];
  int d = dst[e];
  float nrm = dinv[s] * dinv[d];
  float* ap = agg + (size_t)d * 256 + c;
  __builtin_prefetch(ap, 1, 3);  // global_prefetch_b8: pull RMW line toward L2
  float4 hv = *(const float4*)(h + (size_t)s * 256 + c);
  atomicAdd(ap + 0, hv.x * nrm);
  atomicAdd(ap + 1, hv.y * nrm);
  atomicAdd(ap + 2, hv.z * nrm);
  atomicAdd(ap + 3, hv.w * nrm);
}

// act = f16(relu(agg + h*dinv^2 (self loop) + bias))   over [N,256]
__global__ __launch_bounds__(256)
void finalize_kernel(const float* __restrict__ h, const float* __restrict__ agg,
                     const float* __restrict__ dinv, const float* __restrict__ bias,
                     _Float16* __restrict__ act, size_t total) {
  size_t idx = (size_t)blockIdx.x * blockDim.x + threadIdx.x;
  if (idx >= total) return;
  size_t n = idx >> 8;
  int c = (int)(idx & 255);
  float di = dinv[n];
  float v = agg[idx] + h[idx] * di * di + bias[c];
  act[idx] = (_Float16)fmaxf(v, 0.0f);
}

// ------------------------------- GRU gates ----------------------------------

__global__ __launch_bounds__(256)
void gru_gate_kernel(const float* __restrict__ gi, const float* __restrict__ gh,
                     const float* __restrict__ hprev, float* __restrict__ hout,
                     _Float16* __restrict__ hout_h, size_t total) {
  size_t idx = (size_t)blockIdx.x * blockDim.x + threadIdx.x;
  if (idx >= total) return;
  size_t n = idx >> 7;
  int c = (int)(idx & 127);
  const float* gir = gi + n * 384;
  const float* ghr = gh + n * 384;
  float ir = gir[c], iz = gir[128 + c], inn = gir[256 + c];
  float hr = ghr[c], hz = ghr[128 + c], hn  = ghr[256 + c];
  float r = 1.0f / (1.0f + __expf(-(ir + hr)));
  float z = 1.0f / (1.0f + __expf(-(iz + hz)));
  float nc = tanhf(inn + r * hn);
  float hv = (1.0f - z) * nc + z * hprev[idx];
  hout[idx] = hv;
  hout_h[idx] = (_Float16)hv;
}

// ------------------------- TDM B-tile stage (CDNA5) -------------------------
// One TENSOR_LOAD_TO_LDS per block per k-step: 2-D tile of 64 rows (output
// cols) x 32 f16 (K slice) from BT[Nout][K], row stride K. The descriptor's
// LDS padding (interval 16 DWORDs = one 64B row, amount 4 DWORDs) reproduces
// the 40-half LDS row stride, so fragment indexing matches the manual path.
// D# layout per CDNA5 ISA section 8.3/8.4. This toolchain exposes the 6-arg
// builtin: (g0 u32x4, g1 i32x8, g2 i32x4, g3 i32x4, i32x8, i32 cpol); the
// trailing i32x8 group is unused for a plain 2-D descriptor (zero-filled).

#if USE_TDM
__device__ __forceinline__ void tdm_load_btile(unsigned lds_addr, const _Float16* gptr,
                                               unsigned K, unsigned Nout) {
  unsigned long long ga = (unsigned long long)(uintptr_t)gptr;
  v4u g0 = {
      1u,                                               // count=1 (valid user D#)
      lds_addr,                                         // LDS byte address
      (unsigned)(ga & 0xFFFFFFFFu),                     // global_addr[31:0]
      (unsigned)((ga >> 32) & 0x01FFFFFFu) | 0x80000000u  // addr[56:32] | type=2
  };
  v8i g1 = {
      (int)((1u << 16) |            // data_size = 2 bytes
            (1u << 20) |            // pad_enable
            (3u << 22) |            // pad_interval = 16 DWORDs (one 32-half row)
            (3u << 25)),            // pad_amount   = 4 DWORDs  (8-half pad)
      (int)((K & 0xFFFFu) << 16),                       // tensor_dim0[15:0]
      (int)((K >> 16) | ((Nout & 0xFFFFu) << 16)),      // tensor_dim0 hi | dim1 lo
      (int)((Nout >> 16) | (32u << 16)),                // dim1 hi | tile_dim0 = 32
      (int)64,                                          // tile_dim1 = 64, tile_dim2 = 0
      (int)(K & 0xFFFFFFFFu),                           // tensor_dim0_stride lo
      (int)0,                                           // stride0 hi | stride1 lo
      (int)0                                            // stride1 hi
  };
  v4i gz4 = {0, 0, 0, 0};                               // groups 2/3 unused (2-D)
  v8i gz8 = {0, 0, 0, 0, 0, 0, 0, 0};                   // trailing group unused
  __builtin_amdgcn_tensor_load_to_lds(g0, g1, gz4, gz4, gz8, 0);
}
#endif

// ------------------------------- WMMA GEMM ----------------------------------
// C[M,Nout] = A[M,K] (f16, row-major) x B[K,Nout], B given as BT[Nout][K] f16.
// Block: 256 threads = 8 waves; block tile 128x64; wave tile 16x64
// (4 x v_wmma_f32_16x16x32_f16 accumulators), K-loop step 32.
// B tile staged in LDS by the TDM (double buffered; transfer of tile i+1
// overlaps the WMMAs of tile i), A streamed from global with prefetch.

template<bool BIAS, bool RELU, bool W32, bool W16>
__global__ __launch_bounds__(256)
void gemm_wmma_kernel(const _Float16* __restrict__ A, const _Float16* __restrict__ BT,
                      const float* __restrict__ bias, float* __restrict__ outF,
                      _Float16* __restrict__ outH, int M, int K, int Nout) {
#if USE_TDM
  __shared__ _Float16 ldsB[2][64 * 40];
#else
  __shared__ _Float16 ldsB[1][64 * 40];
#endif

  const int tid  = threadIdx.x;
  const int wave = tid >> 5;
  const int lane = tid & 31;
  const int hl   = lane >> 4;   // half-wave selector (ISA 16-bit fragment layout)
  const int l16  = lane & 15;

  const int col0 = blockIdx.y * 64;
  const int row0 = blockIdx.x * 128 + wave * 16;

  // Clamp A row for M-boundary blocks: keeps EXEC all-ones through the WMMAs
  // (ISA requirement); only the epilogue stores are predicated.
  int arow = row0 + l16;
  if (arow >= M) arow = M - 1;
  const _Float16* Ap = A + (size_t)arow * K;

  v8f acc[4];
#pragma unroll
  for (int c = 0; c < 4; ++c)
#pragma unroll
    for (int j = 0; j < 8; ++j) acc[c][j] = 0.0f;

  const int nk = K >> 5;

#if USE_TDM
  const _Float16* Btile = BT + (size_t)col0 * K;   // tile origin: row=col0, k=0
  // Flat LDS pointers carry the wave-relative LDS byte address in addr[31:0]
  // (CDNA5 ISA 10.2 aperture mapping).
  const unsigned lbuf0 = (unsigned)(uintptr_t)&ldsB[0][0];
  const unsigned lbuf1 = (unsigned)(uintptr_t)&ldsB[1][0];
  if (wave == 0) tdm_load_btile(lbuf0, Btile, (unsigned)K, (unsigned)Nout);
#else
  const int bcol = tid >> 2;
  const int bseg = tid & 3;
  const _Float16* Bp = BT + (size_t)(col0 + bcol) * K + bseg * 8;
  _Float16* Lw = &ldsB[0][bcol * 40 + bseg * 8];
#endif

  for (int i = 0; i < nk; ++i) {
    const int k0 = i << 5;

#if USE_TDM
    if (wave == 0) {
      if (i + 1 < nk) {
        // Prefetch tile i+1 into the other buffer, then wait only for tile i.
        tdm_load_btile((i & 1) ? lbuf0 : lbuf1, Btile + k0 + 32,
                       (unsigned)K, (unsigned)Nout);
        __builtin_amdgcn_s_wait_tensorcnt((short)1);
      } else {
        __builtin_amdgcn_s_wait_tensorcnt((short)0);
      }
    }
    __syncthreads();                       // tile i visible to all 8 waves
    const _Float16* lb = &ldsB[i & 1][0];
#else
    __syncthreads();
    *(v8h*)Lw = *(const v8h*)(Bp + k0);    // global b128 -> ds b128
    __syncthreads();
    const _Float16* lb = &ldsB[0][0];
#endif

    if (k0 + 32 < K) __builtin_prefetch(Ap + k0 + 32, 0, 3);

    // A fragment (16-bit 16x32 layout): lane holds K {hl*8..+7} and {16+hl*8..+7}
    v8h a0 = *(const v8h*)(Ap + k0 + hl * 8);
    v8h a1 = *(const v8h*)(Ap + k0 + 16 + hl * 8);
    v16h af;
#pragma unroll
    for (int j = 0; j < 8; ++j) { af[j] = a0[j]; af[j + 8] = a1[j]; }

#pragma unroll
    for (int c = 0; c < 4; ++c) {
      // B fragment (16-bit 32x16 layout): lane holds K {hl*8..+7} and {16+hl*8..+7}
      const _Float16* bp = lb + (c * 16 + l16) * 40 + hl * 8;
      v8h b0 = *(const v8h*)bp;
      v8h b1 = *(const v8h*)(bp + 16);
      v16h bf;
#pragma unroll
      for (int j = 0; j < 8; ++j) { bf[j] = b0[j]; bf[j + 8] = b1[j]; }
      acc[c] = __builtin_amdgcn_wmma_f32_16x16x32_f16(
          /*neg_a=*/false, af, /*neg_b=*/false, bf,
          /*c_mod=*/(short)0, acc[c], /*reuse_a=*/false, /*reuse_b=*/false);
    }

#if USE_TDM
    __syncthreads();   // WAR: all reads of this buffer done before TDM reuses it
#endif
  }

  // Epilogue: C/D layout -> row = row0 + hl*8 + j, col = col0 + c*16 + l16
#pragma unroll
  for (int c = 0; c < 4; ++c) {
    const int col = col0 + c * 16 + l16;
    float bv = 0.0f;
    if constexpr (BIAS) bv = bias[col];
#pragma unroll
    for (int j = 0; j < 8; ++j) {
      const int row = row0 + hl * 8 + j;
      if (row < M) {
        float v = acc[c][j] + bv;
        if constexpr (RELU) v = fmaxf(v, 0.0f);
        if constexpr (W32) outF[(size_t)row * Nout + col] = v;
        if constexpr (W16) outH[(size_t)row * Nout + col] = (_Float16)v;
      }
    }
  }
}

// ---------------------------------------------------------------------------

extern "C" void kernel_launch(void* const* d_in, const int* in_sizes, int n_in,
                              void* d_out, int out_size, void* d_ws, size_t ws_size,
                              hipStream_t stream) {
  (void)n_in; (void)out_size; (void)ws_size;

  const int OBS = 128, HID = 128, HDIM = 256, G3 = 384;
  const int N = in_sizes[0] / OBS;
  const int E = in_sizes[2] / 2;

  const float* obs      = (const float*)d_in[0];
  const float* hid      = (const float*)d_in[1];
  const int*   src      = (const int*)d_in[2];
  const int*   dst      = src + E;
  const float* conv_o_w = (const float*)d_in[3];
  const float* conv_o_b = (const float*)d_in[4];
  const float* fc_o_w1  = (const float*)d_in[5];
  const float* fc_o_b1  = (const float*)d_in[6];
  const float* fc_o_w2  = (const float*)d_in[7];
  const float* fc_o_b2  = (const float*)d_in[8];
  const float* conv_h_w = (const float*)d_in[9];
  const float* conv_h_b = (const float*)d_in[10];
  const float* fc_h_w1  = (const float*)d_in[11];
  const float* fc_h_b1  = (const float*)d_in[12];
  const float* fc_h_w2  = (const float*)d_in[13];
  const float* fc_h_b2  = (const float*)d_in[14];
  const float* gru_w_ih = (const float*)d_in[15];
  const float* gru_w_hh = (const float*)d_in[16];
  const float* gru_b_ih = (const float*)d_in[17];
  const float* gru_b_hh = (const float*)d_in[18];
  const float* enc_w    = (const float*)d_in[19];
  const float* enc_b    = (const float*)d_in[20];

  float* latent   = (float*)d_out;                  // [N, 256]
  float* next_hid = latent + (size_t)N * HDIM;      // [N, 128]

  // ---- workspace arena (256B-aligned bump allocator) ----
  char* ws = (char*)d_ws;
  size_t off = 0;
  auto alloc = [&](size_t bytes) -> void* {
    void* p = ws + off;
    off += (bytes + 255) & ~(size_t)255;
    return p;
  };

  float*    deg     = (float*)alloc((size_t)N * 4);
  float*    dinv    = (float*)alloc((size_t)N * 4);
  _Float16* xo_h    = (_Float16*)alloc((size_t)N * OBS * 2);   // obs f16; reused as phi_h
  _Float16* xh_h    = (_Float16*)alloc((size_t)N * HID * 2);   // hid f16; reused as psi_h
  _Float16* convT_o = (_Float16*)alloc((size_t)HDIM * OBS * 2);
  _Float16* w1T_o   = (_Float16*)alloc((size_t)HDIM * HDIM * 2);
  _Float16* w2T_o   = (_Float16*)alloc((size_t)OBS * HDIM * 2);
  _Float16* convT_h = (_Float16*)alloc((size_t)HDIM * HID * 2);
  _Float16* w1T_h   = (_Float16*)alloc((size_t)HDIM * HDIM * 2);
  _Float16* w2T_h   = (_Float16*)alloc((size_t)HID * HDIM * 2);
  _Float16* wihT    = (_Float16*)alloc((size_t)G3 * OBS * 2);  // already [Nout][K]
  _Float16* whhT    = (_Float16*)alloc((size_t)G3 * HID * 2);  // already [Nout][K]
  _Float16* encT    = (_Float16*)alloc((size_t)HDIM * HID * 2);
  float*    bufA    = (float*)alloc((size_t)N * G3 * 4);   // conv-out [N,256] / gi [N,384]
  float*    bufB    = (float*)alloc((size_t)N * G3 * 4);   // agg      [N,256] / gh [N,384]
  _Float16* actC    = (_Float16*)alloc((size_t)N * HDIM * 2);  // conv act; reused as psi_f32
  _Float16* actD    = (_Float16*)alloc((size_t)N * HDIM * 2);  // fc1 act;  reused as nh f16

  _Float16* phi_h = xo_h;            // free after first GEMM
  _Float16* psi_h = xh_h;            // free after branch-H conv GEMM
  float*    psi_f = (float*)actC;    // free after branch-H fc1 GEMM (N*128 f32 fits)
  _Float16* nh_h  = actD;            // free after branch-H fc2 GEMM

  const dim3 blk(256);
  const int gx = (N + 127) / 128;    // GEMM row blocks
  auto eb = [&](size_t n) { return (unsigned)((n + 255) / 256); };  // elementwise blocks

  // ---- input casts + weight prep (f16, B pre-transposed to [Nout][K]) ----
  cast_f32_f16_kernel<<<eb((size_t)N * OBS), blk, 0, stream>>>(obs, xo_h, (size_t)N * OBS);
  cast_f32_f16_kernel<<<eb((size_t)N * HID), blk, 0, stream>>>(hid, xh_h, (size_t)N * HID);
  transpose_cast_kernel<<<eb(OBS * HDIM), blk, 0, stream>>>(conv_o_w, convT_o, OBS, HDIM);
  transpose_cast_kernel<<<eb(HDIM * HDIM), blk, 0, stream>>>(fc_o_w1, w1T_o, HDIM, HDIM);
  transpose_cast_kernel<<<eb(HDIM * OBS), blk, 0, stream>>>(fc_o_w2, w2T_o, HDIM, OBS);
  transpose_cast_kernel<<<eb(HID * HDIM), blk, 0, stream>>>(conv_h_w, convT_h, HID, HDIM);
  transpose_cast_kernel<<<eb(HDIM * HDIM), blk, 0, stream>>>(fc_h_w1, w1T_h, HDIM, HDIM);
  transpose_cast_kernel<<<eb(HDIM * HID), blk, 0, stream>>>(fc_h_w2, w2T_h, HDIM, HID);
  cast_f32_f16_kernel<<<eb((size_t)G3 * OBS), blk, 0, stream>>>(gru_w_ih, wihT, (size_t)G3 * OBS);
  cast_f32_f16_kernel<<<eb((size_t)G3 * HID), blk, 0, stream>>>(gru_w_hh, whhT, (size_t)G3 * HID);
  transpose_cast_kernel<<<eb(HID * HDIM), blk, 0, stream>>>(enc_w, encT, HID, HDIM);

  // ---- symmetric GCN normalization: dinv = rsqrt(deg_dst + 1) ----
  zero_f32_kernel<<<eb(N), blk, 0, stream>>>(deg, (size_t)N);
  degree_kernel<<<eb(E), blk, 0, stream>>>(dst, deg, E);
  dinv_kernel<<<eb(N), blk, 0, stream>>>(deg, dinv, N);

  const size_t nh = (size_t)N * HDIM;
  const unsigned sb = (unsigned)(((size_t)E * 64 + 255) / 256);

  // ---- branch O: GCNConv(obs) -> relu -> fc1 -> relu -> fc2 = phi ----
  gemm_wmma_kernel<false, false, true, false><<<dim3(gx, HDIM / 64), blk, 0, stream>>>(
      xo_h, convT_o, nullptr, bufA, nullptr, N, OBS, HDIM);
  zero_f32_kernel<<<eb(nh), blk, 0, stream>>>(bufB, nh);
  scatter_kernel<<<sb, blk, 0, stream>>>(src, dst, bufA, dinv, bufB, E);
  finalize_kernel<<<eb(nh), blk, 0, stream>>>(bufA, bufB, dinv, conv_o_b, actC, nh);
  gemm_wmma_kernel<true, true, false, true><<<dim3(gx, HDIM / 64), blk, 0, stream>>>(
      actC, w1T_o, fc_o_b1, nullptr, actD, N, HDIM, HDIM);
  gemm_wmma_kernel<true, false, false, true><<<dim3(gx, OBS / 64), blk, 0, stream>>>(
      actD, w2T_o, fc_o_b2, nullptr, phi_h, N, HDIM, OBS);

  // ---- branch H: GCNConv(hidden) -> relu -> fc1 -> relu -> fc2 = psi ----
  gemm_wmma_kernel<false, false, true, false><<<dim3(gx, HDIM / 64), blk, 0, stream>>>(
      xh_h, convT_h, nullptr, bufA, nullptr, N, HID, HDIM);
  zero_f32_kernel<<<eb(nh), blk, 0, stream>>>(bufB, nh);
  scatter_kernel<<<sb, blk, 0, stream>>>(src, dst, bufA, dinv, bufB, E);
  finalize_kernel<<<eb(nh), blk, 0, stream>>>(bufA, bufB, dinv, conv_h_b, actC, nh);
  gemm_wmma_kernel<true, true, false, true><<<dim3(gx, HDIM / 64), blk, 0, stream>>>(
      actC, w1T_h, fc_h_b1, nullptr, actD, N, HDIM, HDIM);
  // psi needed both as f16 (GRU GEMM input) and f32 (gate combine)
  gemm_wmma_kernel<true, false, true, true><<<dim3(gx, HID / 64), blk, 0, stream>>>(
      actD, w2T_h, fc_h_b2, psi_f, psi_h, N, HDIM, HID);

  // ---- GRU cell ----
  gemm_wmma_kernel<true, false, true, false><<<dim3(gx, G3 / 64), blk, 0, stream>>>(
      phi_h, wihT, gru_b_ih, bufA, nullptr, N, OBS, G3);   // gi = phi @ w_ih.T + b_ih
  gemm_wmma_kernel<true, false, true, false><<<dim3(gx, G3 / 64), blk, 0, stream>>>(
      psi_h, whhT, gru_b_hh, bufB, nullptr, N, HID, G3);   // gh = psi @ w_hh.T + b_hh
  gru_gate_kernel<<<eb((size_t)N * HID), blk, 0, stream>>>(
      bufA, bufB, psi_f, next_hid, nh_h, (size_t)N * HID);

  // ---- encoder: latent = next_hid @ enc_w + enc_b ----
  gemm_wmma_kernel<true, false, true, false><<<dim3(gx, HDIM / 64), blk, 0, stream>>>(
      nh_h, encT, enc_b, latent, nullptr, N, HID, HDIM);
}
